// GINet_40011915329789
// MI455X (gfx1250) — matrix operand order repaired and checked by reading, not compiled, not run on hardware
//
#include <hip/hip_runtime.h>
#include <hip/hip_bf16.h>

typedef __attribute__((ext_vector_type(16))) _Float16 v16h;
typedef __attribute__((ext_vector_type(8)))  _Float16 v8h;
typedef __attribute__((ext_vector_type(8)))  float    v8f;

#define NN   100000
#define EE   200000
#define GG   4096
#define LL   5
#define DD   300
#define FEATD 512
#define EPSBN 1e-5f

// ---------------- utility ----------------
__global__ void k_zero(float* p, long long n) {
  long long i = (long long)blockIdx.x * blockDim.x + threadIdx.x;
  long long s = (long long)gridDim.x * blockDim.x;
  for (; i < n; i += s) p[i] = 0.f;
}

// h[i,d] = x_emb1[x[i,0],d] + x_emb2[x[i,1],d]
__global__ void k_embed(const int* __restrict__ x, const float* __restrict__ e1,
                        const float* __restrict__ e2, float* __restrict__ h) {
  long long i = (long long)blockIdx.x * blockDim.x + threadIdx.x;
  long long total = (long long)NN * DD;
  if (i >= total) return;
  long long node = i / DD; int d = (int)(i % DD);
  h[i] = e1[(long long)x[node * 2 + 0] * DD + d] + e2[(long long)x[node * 2 + 1] * DD + d];
}

__global__ void k_count(const int* __restrict__ batch, float* __restrict__ cnt) {
  int i = blockIdx.x * blockDim.x + threadIdx.x;
  if (i < NN) atomicAdd(&cnt[batch[i]], 1.f);
}

// msg = h[src] + ee1[type] + ee2[dir]; atomically scatter-add into agg[dst]
__global__ void k_scatter(const float* __restrict__ h, const int* __restrict__ edge_index,
                          const int* __restrict__ edge_attr, const float* __restrict__ ee1l,
                          const float* __restrict__ ee2l, float* __restrict__ agg) {
  long long i = (long long)blockIdx.x * blockDim.x + threadIdx.x;
  long long total = (long long)(EE + NN) * DD;
  if (i >= total) return;
  long long e = i / DD; int d = (int)(i % DD);
  int s, t, ty, dr;
  if (e < EE) {
    s = edge_index[e]; t = edge_index[EE + e];
    ty = edge_attr[e * 2 + 0]; dr = edge_attr[e * 2 + 1];
  } else {
    s = t = (int)(e - EE); ty = 4; dr = 0;  // self loops
  }
  float v = h[(long long)s * DD + d] + ee1l[ty * DD + d] + ee2l[dr * DD + d];
  atomicAdd(&agg[(long long)t * DD + d], v);
}

// ---------------- GEMM staging: fp32 -> zero-padded f16 ----------------
// A16[Mp, Kp] row-major, zero padded
__global__ void k_cvt_a16(const float* __restrict__ A, _Float16* __restrict__ A16,
                          int M, int K, int Kp, long long total) {
  long long i = (long long)blockIdx.x * blockDim.x + threadIdx.x;
  if (i >= total) return;
  long long row = i / Kp; int k = (int)(i % Kp);
  float v = (row < M && k < K) ? A[row * (long long)K + k] : 0.f;
  A16[i] = (_Float16)v;
}

// Pack B[K,N] into per-lane WMMA B-fragment order:
// Bp[(ct*Kt + kt)*512 + lane*16 + i] = B[kt*32 + (i>>3)*16 + (lane>>4)*8 + (i&7)][ct*16 + (lane&15)]
__global__ void k_pack_b(const float* __restrict__ B, _Float16* __restrict__ Bp,
                         int K, int N, int Kt, long long total) {
  long long i = (long long)blockIdx.x * blockDim.x + threadIdx.x;
  if (i >= total) return;
  int j  = (int)(i & 511);
  long long t = i >> 9;                 // ct*Kt + kt
  int kt = (int)(t % Kt);
  int ct = (int)(t / Kt);
  int L = j >> 4, ii = j & 15;
  int k = kt * 32 + (ii >> 3) * 16 + (L >> 4) * 8 + (ii & 7);
  int n = ct * 16 + (L & 15);
  float v = (k < K && n < N) ? B[(long long)k * N + n] : 0.f;
  Bp[i] = (_Float16)v;
}

// ---------------- WMMA GEMM: guard-free streaming inner loop ----------------
// block = 256 (8 waves); wave tile = 32 rows x 64 cols (2x4 accumulators);
// block tile = 256 x 64. A16 padded [Mp,Kp]; Bp pre-packed fragments.
__global__ void k_gemm_wmma(const _Float16* __restrict__ A16, const _Float16* __restrict__ Bp,
                            const float* __restrict__ bias, float* __restrict__ C,
                            int M, int N, int Kt, int Kp, int relu) {
  const int lane = threadIdx.x & 31;
  const int wave = threadIdx.x >> 5;
  const int m  = lane & 15;
  const int kh = lane >> 4;
  const long long rowBase = ((long long)blockIdx.x * 8 + wave) * 32;
  const int ct0 = blockIdx.y * 4;
  const _Float16* arow0 = A16 + (rowBase + m) * (long long)Kp;
  const _Float16* arow1 = arow0 + 16 * (long long)Kp;
  const _Float16* bbase = Bp + (long long)ct0 * Kt * 512 + lane * 16;
  const long long bstride = (long long)Kt * 512;

  v8f acc[2][4] = {{v8f{}, v8f{}, v8f{}, v8f{}}, {v8f{}, v8f{}, v8f{}, v8f{}}};
  for (int kt = 0; kt < Kt; ++kt) {
    const int k0 = kt << 5;
    if (kt + 1 < Kt) {
      __builtin_prefetch(arow0 + k0 + 32, 0, 1);
      __builtin_prefetch(arow1 + k0 + 32, 0, 1);
    }
    // A fragments: contiguous 16B loads per ISA 16-bit 16x32 layout
    v8h a0lo = *(const v8h*)(arow0 + k0 + kh * 8);
    v8h a0hi = *(const v8h*)(arow0 + k0 + 16 + kh * 8);
    v8h a1lo = *(const v8h*)(arow1 + k0 + kh * 8);
    v8h a1hi = *(const v8h*)(arow1 + k0 + 16 + kh * 8);
    v16h a0, a1;
#pragma unroll
    for (int i = 0; i < 8; ++i) {
      a0[i] = a0lo[i]; a0[i + 8] = a0hi[i];
      a1[i] = a1lo[i]; a1[i + 8] = a1hi[i];
    }
    const _Float16* bk = bbase + (long long)kt * 512;
    v16h b0 = *(const v16h*)(bk);
    v16h b1 = *(const v16h*)(bk + bstride);
    v16h b2 = *(const v16h*)(bk + 2 * bstride);
    v16h b3 = *(const v16h*)(bk + 3 * bstride);
    acc[0][0] = __builtin_amdgcn_wmma_f32_16x16x32_f16(false, a0, false, b0, (short)0, acc[0][0], false, false);
    acc[0][1] = __builtin_amdgcn_wmma_f32_16x16x32_f16(false, a0, false, b1, (short)0, acc[0][1], false, false);
    acc[0][2] = __builtin_amdgcn_wmma_f32_16x16x32_f16(false, a0, false, b2, (short)0, acc[0][2], false, false);
    acc[0][3] = __builtin_amdgcn_wmma_f32_16x16x32_f16(false, a0, false, b3, (short)0, acc[0][3], false, false);
    acc[1][0] = __builtin_amdgcn_wmma_f32_16x16x32_f16(false, a1, false, b0, (short)0, acc[1][0], false, false);
    acc[1][1] = __builtin_amdgcn_wmma_f32_16x16x32_f16(false, a1, false, b1, (short)0, acc[1][1], false, false);
    acc[1][2] = __builtin_amdgcn_wmma_f32_16x16x32_f16(false, a1, false, b2, (short)0, acc[1][2], false, false);
    acc[1][3] = __builtin_amdgcn_wmma_f32_16x16x32_f16(false, a1, false, b3, (short)0, acc[1][3], false, false);
  }
  // store: C/D layout — VGPR r: lanes 0-15 M=r, lanes 16-31 M=r+8; col = lane&15
#pragma unroll
  for (int c = 0; c < 4; ++c) {
    int col = (ct0 + c) * 16 + m;
    if (col < N) {
      float bv = bias ? bias[col] : 0.f;
#pragma unroll
      for (int t = 0; t < 2; ++t) {
#pragma unroll
        for (int r = 0; r < 8; ++r) {
          long long row = rowBase + t * 16 + kh * 8 + r;
          if (row < M) {
            float v = acc[t][c][r] + bv;
            if (relu) v = fmaxf(v, 0.f);
            C[row * (long long)N + col] = v;
          }
        }
      }
    }
  }
}

// per-column sum & sumsq (strip-mined, few atomics)
__global__ void k_colreduce(const float* __restrict__ Z, float* __restrict__ stats,
                            int M, int C) {
  int c = blockIdx.x * blockDim.x + threadIdx.x;
  if (c >= C) return;
  float s = 0.f, sq = 0.f;
  for (long long r = blockIdx.y; r < M; r += gridDim.y) {
    float v = Z[r * (long long)C + c];
    s += v; sq += v * v;
  }
  atomicAdd(&stats[c], s);
  atomicAdd(&stats[C + c], sq);
}

// act: 0=none, 1=relu, 2=softplus
__global__ void k_bn_apply(const float* __restrict__ Zin, float* __restrict__ Zout,
                           const float* __restrict__ stats, const float* __restrict__ g,
                           const float* __restrict__ b, int M, int C, int act) {
  long long i = (long long)blockIdx.x * blockDim.x + threadIdx.x;
  long long total = (long long)M * C;
  if (i >= total) return;
  int c = (int)(i % C);
  float mean = stats[c] / (float)M;
  float var  = stats[C + c] / (float)M - mean * mean;
  float v = (Zin[i] - mean) * rsqrtf(var + EPSBN) * g[c] + b[c];
  if (act == 1)      v = fmaxf(v, 0.f);
  else if (act == 2) v = fmaxf(v, 0.f) + log1pf(expf(-fabsf(v)));  // stable softplus
  Zout[i] = v;
}

__global__ void k_pool(const float* __restrict__ h, const int* __restrict__ batch,
                       float* __restrict__ pooled, int l) {
  long long i = (long long)blockIdx.x * blockDim.x + threadIdx.x;
  long long total = (long long)NN * DD;
  if (i >= total) return;
  long long node = i / DD; int d = (int)(i % DD);
  atomicAdd(&pooled[(long long)batch[node] * (LL * DD) + l * DD + d], h[i]);
}

__global__ void k_pooldiv(float* __restrict__ pooled, const float* __restrict__ cnt) {
  long long i = (long long)blockIdx.x * blockDim.x + threadIdx.x;
  long long total = (long long)GG * LL * DD;
  if (i >= total) return;
  long long g = i / (LL * DD);
  pooled[i] /= fmaxf(cnt[g], 1.f);
}

// pred = p[G,256] @ Wp2[256,2] + bp2
__global__ void k_pred(const float* __restrict__ p, const float* __restrict__ W,
                       const float* __restrict__ bias, float* __restrict__ out) {
  int i = blockIdx.x * blockDim.x + threadIdx.x;
  if (i >= GG * 2) return;
  int g = i >> 1, j = i & 1;
  float s = bias[j];
  for (int k = 0; k < FEATD / 2; ++k) s += p[g * (FEATD / 2) + k] * W[k * 2 + j];
  out[i] = s;
}

static inline long long cdiv(long long a, long long b) { return (a + b - 1) / b; }

static void run_gemm(const float* A, const float* B, const float* bias, float* C,
                     int M, int K, int N, int relu,
                     _Float16* A16, _Float16* Bp, hipStream_t stream) {
  const int BT = 256;
  int Mp = (int)cdiv(M, 256) * 256;
  int Kp = (int)cdiv(K, 32) * 32, Kt = Kp / 32;
  int Np = (int)cdiv(N, 64) * 64, Ct = Np / 16;
  long long atot = (long long)Mp * Kp;
  k_cvt_a16<<<(unsigned)cdiv(atot, BT), BT, 0, stream>>>(A, A16, M, K, Kp, atot);
  long long btot = (long long)Ct * Kt * 512;
  k_pack_b<<<(unsigned)cdiv(btot, BT), BT, 0, stream>>>(B, Bp, K, N, Kt, btot);
  dim3 grid((unsigned)(Mp / 256), (unsigned)(Np / 64));
  k_gemm_wmma<<<grid, BT, 0, stream>>>(A16, Bp, bias, C, M, N, Kt, Kp, relu);
}

extern "C" void kernel_launch(void* const* d_in, const int* in_sizes, int n_in,
                              void* d_out, int out_size, void* d_ws, size_t ws_size,
                              hipStream_t stream) {
  const int*   x          = (const int*)d_in[0];
  const int*   edge_index = (const int*)d_in[1];
  const int*   edge_attr  = (const int*)d_in[2];
  const int*   batch      = (const int*)d_in[3];
  const float* x_emb1     = (const float*)d_in[4];
  const float* x_emb2     = (const float*)d_in[5];
  const float* ee1        = (const float*)d_in[6];
  const float* ee2        = (const float*)d_in[7];
  const float* W1         = (const float*)d_in[8];
  const float* b1         = (const float*)d_in[9];
  const float* g1         = (const float*)d_in[10];
  const float* be1        = (const float*)d_in[11];
  const float* W2         = (const float*)d_in[12];
  const float* b2         = (const float*)d_in[13];
  const float* g_bn       = (const float*)d_in[14];
  const float* b_bn       = (const float*)d_in[15];
  const float* Wf         = (const float*)d_in[16];
  const float* bf         = (const float*)d_in[17];
  const float* Wp1        = (const float*)d_in[18];
  const float* bp1        = (const float*)d_in[19];
  const float* gp         = (const float*)d_in[20];
  const float* bp         = (const float*)d_in[21];
  const float* Wp2        = (const float*)d_in[22];
  const float* bp2        = (const float*)d_in[23];

  float* ws = (float*)d_ws;
  const long long nD = (long long)NN * DD;          // 30M
  float* h    = ws;                                  // [N,300]
  float* agg  = ws + nD;                             // [N,300] (reused as z2)
  float* z    = ws + 2 * nD;                         // [N,600]
  float* hexp = ws + 4 * nD;                         // [G,512]
  float* pbuf = hexp + (long long)GG * FEATD;        // [G,256]
  float* cnt  = pbuf + (long long)GG * (FEATD / 2);  // [G]
  float* stats = cnt + GG;                           // [1200]
  long long off16 = (stats + 1200) - ws;
  off16 = (off16 + 3) & ~3LL;                        // 16B align
  _Float16* A16 = (_Float16*)(ws + off16);           // up to 100096*608 f16
  _Float16* Bp  = A16 + (long long)100096 * 608;     // up to ~1M f16 packed weights

  float* pooled = (float*)d_out;                     // [G, L*D] = h_global
  float* pred   = pooled + (long long)GG * LL * DD;  // [G, 2]

  const int B = 256;

  // init
  k_zero<<<(unsigned)cdiv((long long)GG * LL * DD, B), B, 0, stream>>>(pooled, (long long)GG * LL * DD);
  k_zero<<<(unsigned)cdiv(GG, B), B, 0, stream>>>(cnt, GG);
  k_count<<<(unsigned)cdiv(NN, B), B, 0, stream>>>(batch, cnt);
  k_embed<<<(unsigned)cdiv(nD, B), B, 0, stream>>>(x, x_emb1, x_emb2, h);

  for (int l = 0; l < LL; ++l) {
    // message passing: agg = segment_sum(h[src] + eemb, dst)
    k_zero<<<(unsigned)cdiv(nD, B), B, 0, stream>>>(agg, nD);
    k_scatter<<<(unsigned)cdiv((long long)(EE + NN) * DD, B), B, 0, stream>>>(
        h, edge_index, edge_attr, ee1 + (long long)l * 5 * DD, ee2 + (long long)l * 3 * DD, agg);

    // z = agg @ W1[l] + b1[l]   [N,600]
    run_gemm(agg, W1 + (long long)l * DD * 2 * DD, b1 + (long long)l * 2 * DD, z,
             NN, DD, 2 * DD, 0, A16, Bp, stream);

    // BN + relu over N
    k_zero<<<(unsigned)cdiv(2 * 2 * DD, B), B, 0, stream>>>(stats, 2 * 2 * DD);
    { dim3 grid((unsigned)cdiv(2 * DD, B), 256);
      k_colreduce<<<grid, B, 0, stream>>>(z, stats, NN, 2 * DD); }
    k_bn_apply<<<(unsigned)cdiv((long long)NN * 2 * DD, B), B, 0, stream>>>(
        z, z, stats, g1 + (long long)l * 2 * DD, be1 + (long long)l * 2 * DD, NN, 2 * DD, 1);

    // z2 = relu(z @ W2[l] + b2[l])   [N,300] -> reuse agg
    run_gemm(z, W2 + (long long)l * 2 * DD * DD, b2 + (long long)l * DD, agg,
             NN, 2 * DD, DD, 1, A16, Bp, stream);

    // outer BN (+relu except last layer); write new h
    k_zero<<<(unsigned)cdiv(2 * DD, B), B, 0, stream>>>(stats, 2 * DD);
    { dim3 grid((unsigned)cdiv(DD, B), 256);
      k_colreduce<<<grid, B, 0, stream>>>(agg, stats, NN, DD); }
    k_bn_apply<<<(unsigned)cdiv(nD, B), B, 0, stream>>>(
        agg, h, stats, g_bn + (long long)l * DD, b_bn + (long long)l * DD, NN, DD,
        (l < LL - 1) ? 1 : 0);

    // pooled[:, l*D:(l+1)*D] += segment_sum(h, batch)
    k_pool<<<(unsigned)cdiv(nD, B), B, 0, stream>>>(h, batch, pooled, l);
  }

  // mean pooling -> h_global (in d_out)
  k_pooldiv<<<(unsigned)cdiv((long long)GG * LL * DD, B), B, 0, stream>>>(pooled, cnt);

  // h_exp = h_global @ Wf + bf   [4096,1500]x[1500,512]
  run_gemm(pooled, Wf, bf, hexp, GG, LL * DD, FEATD, 0, A16, Bp, stream);

  // p_lin = h_exp @ Wp1 + bp1   [4096,512]x[512,256]
  run_gemm(hexp, Wp1, bp1, pbuf, GG, FEATD, FEATD / 2, 0, A16, Bp, stream);

  // softplus(BN(p_lin)) over G rows
  k_zero<<<(unsigned)cdiv(FEATD, B), B, 0, stream>>>(stats, FEATD);
  { dim3 grid((unsigned)cdiv(FEATD / 2, B), 256);
    k_colreduce<<<grid, B, 0, stream>>>(pbuf, stats, GG, FEATD / 2); }
  k_bn_apply<<<(unsigned)cdiv((long long)GG * (FEATD / 2), B), B, 0, stream>>>(
      pbuf, pbuf, stats, gp, bp, GG, FEATD / 2, 2);

  // pred = p @ Wp2 + bp2
  k_pred<<<(unsigned)cdiv(GG * 2, B), B, 0, stream>>>(pbuf, Wp2, bp2, pred);
}
